// MultiHeadedAttention_89438398972419
// MI455X (gfx1250) — compile-verified, matrix operand order checked by reading, not run on hardware
//
#include <hip/hip_runtime.h>
#include <hip/hip_bf16.h>

// MI455X / gfx1250, wave32, WMMA bf16 path.
// Pipeline: cvt(fp32->bf16) -> QKV GEMM (WMMA + TDM-staged B tiles) ->
//           RoPE+transpose -> flash attention (WMMA QK^T + online softmax +
//           WMMA PV, LDS staging) -> dense GEMM (WMMA + TDM) -> fp32 out.

typedef __attribute__((ext_vector_type(16))) __bf16 v16bf;
typedef __attribute__((ext_vector_type(8)))  __bf16 v8bf;
typedef __attribute__((ext_vector_type(8)))  float  v8f;
typedef unsigned int u32x4 __attribute__((ext_vector_type(4)));
typedef int          i32x4 __attribute__((ext_vector_type(4)));
typedef int          i32x8 __attribute__((ext_vector_type(8)));

#define DEVINL __device__ __forceinline__

constexpr int Bc = 2, Sc = 2048, Dc = 1024, Hc = 16, HDc = 64;
#define AW 4  // waves per block in attention kernel

#if __has_builtin(__builtin_amdgcn_tensor_load_to_lds)
#define HAVE_TDM 1
#else
#define HAVE_TDM 0
#endif

// ---- WMMA fragment helpers (ISA 7.12.2 wave32 layouts) ---------------------
// A fragment, 16x32 bf16, row-major source [rows, ld]:
//   lanes 0-15 : row = row0+lane,    K = {k0..k0+7} and {k0+16..k0+23}
//   lanes 16-31: row = row0+lane-16, K = {k0+8..k0+15} and {k0+24..k0+31}
// B fragment (column n of B = source row n of a row-major [N,K] matrix)
// has the identical per-lane pattern, so both use the same loader.
DEVINL const __bf16* frag_ptr(const __bf16* base, int row0, int ld, int k0, int lane) {
  return base + (size_t)(row0 + (lane & 15)) * (size_t)ld
              + (size_t)(k0 + ((lane >> 4) << 3));
}

DEVINL v16bf load_frag(const __bf16* base, int row0, int ld, int k0, int lane) {
  const __bf16* p = frag_ptr(base, row0, ld, k0, lane);
  v8bf lo = *(const v8bf*)p;
  v8bf hi = *(const v8bf*)(p + 16);
  v16bf r;
#pragma unroll
  for (int i = 0; i < 8; ++i) { r[i] = lo[i]; r[i + 8] = hi[i]; }
  return r;
}

DEVINL v8f wmma_bf16(v16bf a, v16bf b, v8f c) {
  return __builtin_amdgcn_wmma_f32_16x16x32_bf16(
      /*neg_a=*/false, a, /*neg_b=*/false, b,
      /*c_mod=*/(short)0, c, /*reuse_a=*/false, /*reuse_b=*/false);
}

// ---- Tensor Data Mover: 2-D tile (tile_d0 x tile_d1, 2-byte elems) -> LDS --
// D# per ISA 8.3/8.4: group0 {count=1, lds_addr, global_addr, type=2},
// group1 {data_size=1(2B), tensor_dim0/1, tile_dim0/1, tensor_dim0_stride}.
#if HAVE_TDM
DEVINL void tdm_load_2d_to_lds(unsigned lds_addr, const void* gsrc,
                               unsigned tensor_d0, unsigned tensor_d1,
                               unsigned stride_d0, unsigned tile_d0,
                               unsigned tile_d1) {
  const unsigned long long ga = (unsigned long long)gsrc;
  u32x4 g0;
  g0[0] = 1u;                                              // count=1 (valid D#)
  g0[1] = lds_addr;                                        // LDS byte address
  g0[2] = (unsigned)ga;                                    // global_addr[31:0]
  g0[3] = (unsigned)((ga >> 32) & 0x01FFFFFFull) | 0x80000000u;  // type=2
  i32x8 g1;
  g1[0] = (int)(1u << 16);                                 // data_size=1 -> 2B
  g1[1] = (int)((tensor_d0 & 0xFFFFu) << 16);              // tensor_dim0 lo16
  g1[2] = (int)((tensor_d0 >> 16) | ((tensor_d1 & 0xFFFFu) << 16));
  g1[3] = (int)((tensor_d1 >> 16) | (tile_d0 << 16));      // tile_dim0
  g1[4] = (int)(tile_d1 & 0xFFFFu);                        // tile_dim1 (dim2=0)
  g1[5] = (int)stride_d0;                                  // dim0_stride lo32
  g1[6] = 0;
  g1[7] = 0;
  i32x4 z4 = {};
#if defined(__clang_major__) && (__clang_major__ >= 23)
  i32x8 z8 = {};
  __builtin_amdgcn_tensor_load_to_lds(g0, g1, z4, z4, z8, 0);
#else
  __builtin_amdgcn_tensor_load_to_lds(g0, g1, z4, z4, 0);
#endif
}
#endif

// ---- fp32 -> bf16 conversion ----------------------------------------------
__global__ __launch_bounds__(256) void cvt_f32_bf16(const float* __restrict__ in,
                                                    __bf16* __restrict__ out, int n4) {
  int i = blockIdx.x * blockDim.x + threadIdx.x;
  if (i >= n4) return;
  float4 v = ((const float4*)in)[i];
  __bf16* op = out + 4 * (size_t)i;
  op[0] = (__bf16)v.x; op[1] = (__bf16)v.y;
  op[2] = (__bf16)v.z; op[3] = (__bf16)v.w;
}

// ---- GEMM: C[M,N] = A[M,K] * Bm[N,K]^T + bias ------------------------------
// Block = 8 waves = 128(M) x 64(N) tile. B chunk (64 cols x 64 K, 8 KB) is
// staged to LDS by the TDM, double buffered; tensor loads overlap WMMA on the
// other buffer. Each wave computes 16x64 from its A rows (direct global).
__global__ __launch_bounds__(256) void gemm_bf16_wmma(
    const __bf16* __restrict__ A, const __bf16* __restrict__ Bm,
    const float* __restrict__ bias, float* __restrict__ Cf,
    __bf16* __restrict__ Cb, int M, int N, int K) {
  __shared__ alignas(16) __bf16 sB[2][64 * 64];
  const int lane = threadIdx.x & 31;
  const int w = threadIdx.x >> 5;
  const int nb = N >> 6;
  const int bm = blockIdx.x / nb;
  const int bn = blockIdx.x - bm * nb;
  const int row0 = (bm << 7) + (w << 4);
  const int col0 = bn << 6;

  v8f acc[4] = {};
#if HAVE_TDM
  if (w == 0)  // one TDM issue per workgroup (per-wave op, EXEC ignored)
    tdm_load_2d_to_lds((unsigned)(unsigned long long)&sB[0][0],
                       Bm + (size_t)col0 * K, K, N, K, 64, 64);
#endif
  int cur = 0;
  for (int kc = 0; kc < K; kc += 64) {
#if HAVE_TDM
    if (w == 0) {
      if (kc + 64 < K) {  // issue next chunk, then wait for current (in-order)
        tdm_load_2d_to_lds((unsigned)(unsigned long long)&sB[cur ^ 1][0],
                           Bm + (size_t)col0 * K + (kc + 64), K, N, K, 64, 64);
        __builtin_amdgcn_s_wait_tensorcnt(1);
      } else {
        __builtin_amdgcn_s_wait_tensorcnt(0);
      }
    }
    __syncthreads();  // B chunk `cur` visible to all waves
#else
    {  // cooperative fallback copy (8 KB per chunk)
      const __bf16* src = Bm + (size_t)col0 * K + kc;
      for (int t = threadIdx.x; t < 64 * 8; t += 256) {
        const int r = t >> 3, c8 = (t & 7) << 3;
        *(v8bf*)&sB[cur][r * 64 + c8] = *(const v8bf*)&src[(size_t)r * K + c8];
      }
      __syncthreads();
    }
#endif
    if (kc + 64 < K)  // near (WGP) prefetch of next A chunk
      __builtin_prefetch(frag_ptr(A, row0, K, kc + 64, lane), 0, 3);
#pragma unroll
    for (int kk = 0; kk < 64; kk += 32) {
      v16bf a = load_frag(A, row0, K, kc + kk, lane);
#pragma unroll
      for (int t = 0; t < 4; ++t) {
        v16bf b = load_frag(&sB[cur][0], t << 4, 64, kk, lane);
        acc[t] = wmma_bf16(a, b, acc[t]);
      }
    }
    __syncthreads();  // everyone done reading buffer `cur` before reuse
    cur ^= 1;
  }
  // D layout: VGPR r -> row rb+r, column = col0 + t*16 + (lane&15)
  const int cl = lane & 15;
  const int rb = (lane >> 4) << 3;
#pragma unroll
  for (int t = 0; t < 4; ++t) {
    const int n = col0 + (t << 4) + cl;
    const float bv = bias[n];
#pragma unroll
    for (int r = 0; r < 8; ++r) {
      const float o = acc[t][r] + bv;
      if (Cf) Cf[(size_t)(row0 + rb + r) * N + n] = o;
      else    Cb[(size_t)(row0 + rb + r) * N + n] = (__bf16)o;
    }
  }
}

// ---- RoPE + transpose ------------------------------------------------------
// qkv bf16 [B,S,3*D] (N index = c*1024 + h*64 + d) ->
//   qt [B,H,S,HD] (scaled by 1/sqrt(HD)), kt [B,H,S,HD], vt [B,H,HD,S]
__global__ __launch_bounds__(256) void rope_split(
    const __bf16* __restrict__ qkv, const float* __restrict__ fc,
    const float* __restrict__ fs, __bf16* __restrict__ qt,
    __bf16* __restrict__ kt, __bf16* __restrict__ vt) {
  const int idx = blockIdx.x * blockDim.x + threadIdx.x;  // B*S*H*32
  if (idx >= Bc * Sc * Hc * (HDc / 2)) return;
  const int i = idx & 31;
  const int h = (idx >> 5) & (Hc - 1);
  const int s = (idx >> 9) & (Sc - 1);
  const int b = idx >> 20;

  const size_t rb = (size_t)(b * Sc + s) * (3 * Dc);
  const int col = h * HDc + 2 * i;
  const float c  = fc[s * 32 + i];
  const float sn = fs[s * 32 + i];

  const float q0v = (float)qkv[rb + col],          q1v = (float)qkv[rb + col + 1];
  const float k0v = (float)qkv[rb + Dc + col],     k1v = (float)qkv[rb + Dc + col + 1];
  const float v0v = (float)qkv[rb + 2 * Dc + col], v1v = (float)qkv[rb + 2 * Dc + col + 1];

  const float qs = 0.125f;  // 1/sqrt(64) folded into Q
  const size_t qkb = ((size_t)(b * Hc + h) * Sc + s) * HDc + 2 * i;
  qt[qkb]     = (__bf16)((q0v * c - q1v * sn) * qs);
  qt[qkb + 1] = (__bf16)((q0v * sn + q1v * c) * qs);
  kt[qkb]     = (__bf16)(k0v * c - k1v * sn);
  kt[qkb + 1] = (__bf16)(k0v * sn + k1v * c);

  const size_t vb = (size_t)(b * Hc + h) * HDc * Sc + (size_t)(2 * i) * Sc + s;
  vt[vb]      = (__bf16)v0v;
  vt[vb + Sc] = (__bf16)v1v;
}

// ---- Flash attention: one wave owns a 16-row Q tile ------------------------
__global__ __launch_bounds__(32 * AW) void attn_fwd_wmma(
    const __bf16* __restrict__ qt, const __bf16* __restrict__ kt,
    const __bf16* __restrict__ vt, __bf16* __restrict__ ctx) {
  __shared__ alignas(16) float  sS[AW][16 * 36];  // 16x32 score tile, pad->36
  __shared__ alignas(16) __bf16 sP[AW][16 * 40];  // 16x32 P tile,     pad->40
  __shared__ float sRow[AW][16];                  // per-row scale / 1/l

  const int lane = threadIdx.x & 31;
  const int w = threadIdx.x >> 5;
  const int gw = blockIdx.x * AW + w;
  const int qtile = gw & (Sc / 16 - 1);
  const int bh = gw >> 7;                         // (b*H + h)
  const int q0 = qtile << 4;

  const __bf16* Q  = qt + (size_t)bh * Sc * HDc;
  const __bf16* Kp = kt + (size_t)bh * Sc * HDc;
  const __bf16* Vp = vt + (size_t)bh * HDc * Sc;
  float*  sSw = sS[w];
  __bf16* sPw = sP[w];
  float*  sRw = sRow[w];

  const v16bf qa0 = load_frag(Q, q0, HDc, 0, lane);
  const v16bf qa1 = load_frag(Q, q0, HDc, 32, lane);

  v8f O[4] = {};                 // 16 x 64 fp32 accumulator
  float m_i = -1e30f, l_i = 0.f; // row stats live in lanes 0..15 (row = lane)
  const int cl = lane & 15;
  const int rb = (lane >> 4) << 3;
  const int kend = q0 + 16;      // causal: keys < q0+16

  for (int j = 0; j < kend; j += 32) {
    // S = Q * K^T  (two 16x16 tiles, K-dim = HD = 64 -> 2 wmma each)
    v8f s0 = {}, s1 = {};
    s0 = wmma_bf16(qa0, load_frag(Kp, j,      HDc, 0,  lane), s0);
    s0 = wmma_bf16(qa1, load_frag(Kp, j,      HDc, 32, lane), s0);
    s1 = wmma_bf16(qa0, load_frag(Kp, j + 16, HDc, 0,  lane), s1);
    s1 = wmma_bf16(qa1, load_frag(Kp, j + 16, HDc, 32, lane), s1);

    // stage D-fragment scores to LDS (row-major) for softmax + transpose
#pragma unroll
    for (int r = 0; r < 8; ++r) {
      sSw[(rb + r) * 36 + cl]      = s0[r];
      sSw[(rb + r) * 36 + 16 + cl] = s1[r];
    }
    asm volatile("s_wait_dscnt 0" ::: "memory");

    // online softmax: lane m (< 16) owns query row q0+m
    if (lane < 16) {
      const int row = q0 + lane;
      float sv[32];
      float mx = m_i;
#pragma unroll
      for (int c = 0; c < 32; ++c) {
        const float v = ((j + c) <= row) ? sSw[lane * 36 + c] : -1e30f;
        sv[c] = v;
        mx = fmaxf(mx, v);
      }
      const float scl = __expf(m_i - mx);
      float sum = 0.f;
#pragma unroll
      for (int c = 0; c < 32; ++c) {
        const float p = __expf(sv[c] - mx);
        sum += p;
        sPw[lane * 40 + c] = (__bf16)p;
      }
      l_i = l_i * scl + sum;
      m_i = mx;
      sRw[lane] = scl;
    }
    asm volatile("s_wait_dscnt 0" ::: "memory");

    // rescale accumulator rows by exp(m_old - m_new)
#pragma unroll
    for (int r = 0; r < 8; ++r) {
      const float sc = sRw[rb + r];
#pragma unroll
      for (int t = 0; t < 4; ++t) O[t][r] *= sc;
    }

    // O += P * V  (P: 16x32 A-fragment from LDS, V^T rows as B-fragments)
    const v16bf pa = load_frag(sPw, 0, 40, 0, lane);
#pragma unroll
    for (int t = 0; t < 4; ++t)
      O[t] = wmma_bf16(pa, load_frag(Vp, t << 4, Sc, j, lane), O[t]);
  }

  if (lane < 16) sRw[lane] = 1.0f / l_i;
  asm volatile("s_wait_dscnt 0" ::: "memory");

  // write ctx as bf16 [B,S,D] for the dense GEMM
  const int b = bh >> 4, h = bh & 15;
#pragma unroll
  for (int t = 0; t < 4; ++t) {
#pragma unroll
    for (int r = 0; r < 8; ++r) {
      const float o = O[t][r] * sRw[rb + r];
      ctx[(size_t)(b * Sc + q0 + rb + r) * Dc + h * HDc + (t << 4) + cl] = (__bf16)o;
    }
  }
}

// ---- host-side launch ------------------------------------------------------
extern "C" void kernel_launch(void* const* d_in, const int* in_sizes, int n_in,
                              void* d_out, int out_size, void* d_ws, size_t ws_size,
                              hipStream_t stream) {
  (void)in_sizes; (void)n_in; (void)out_size; (void)ws_size;
  const float* x       = (const float*)d_in[0];
  const float* w_qkv   = (const float*)d_in[1];
  const float* b_qkv   = (const float*)d_in[2];
  const float* w_dense = (const float*)d_in[3];
  const float* b_dense = (const float*)d_in[4];
  const float* fcos    = (const float*)d_in[5];
  const float* fsin    = (const float*)d_in[6];
  float* out = (float*)d_out;

  char* ws = (char*)d_ws;
  size_t off = 0;
  auto carve = [&](size_t bytes) -> void* {
    void* p = ws + off;
    off += (bytes + 255) & ~(size_t)255;
    return p;
  };
  const int M = Bc * Sc;  // 4096
  __bf16* xb   = (__bf16*)carve((size_t)M * Dc * 2);          // 8 MB
  __bf16* wqb  = (__bf16*)carve((size_t)3 * Dc * Dc * 2);     // 6 MB
  __bf16* wdb  = (__bf16*)carve((size_t)Dc * Dc * 2);         // 2 MB
  __bf16* qkvb = (__bf16*)carve((size_t)M * 3 * Dc * 2);      // 24 MB
  __bf16* qtb  = (__bf16*)carve((size_t)M * Dc * 2);          // 8 MB
  __bf16* ktb  = (__bf16*)carve((size_t)M * Dc * 2);          // 8 MB
  __bf16* vtb  = (__bf16*)carve((size_t)M * Dc * 2);          // 8 MB
  __bf16* ctxb = qkvb;  // qkv is dead after rope_split; reuse for ctx

  // fp32 -> bf16 converts
  cvt_f32_bf16<<<(M * Dc / 4 + 255) / 256, 256, 0, stream>>>(x, xb, M * Dc / 4);
  cvt_f32_bf16<<<(3 * Dc * Dc / 4 + 255) / 256, 256, 0, stream>>>(w_qkv, wqb, 3 * Dc * Dc / 4);
  cvt_f32_bf16<<<(Dc * Dc / 4 + 255) / 256, 256, 0, stream>>>(w_dense, wdb, Dc * Dc / 4);

  // QKV GEMM: [4096,1024] x [1024,3072] + bias -> bf16
  {
    const int blocks = (M / 128) * ((3 * Dc) / 64);  // 32*48 = 1536
    gemm_bf16_wmma<<<blocks, 256, 0, stream>>>(xb, wqb, b_qkv,
                                               nullptr, qkvb, M, 3 * Dc, Dc);
  }
  // RoPE + head transpose (+ V transposed to [B,H,HD,S])
  {
    const int tot = Bc * Sc * Hc * (HDc / 2);        // 2M threads
    rope_split<<<tot / 256, 256, 0, stream>>>(qkvb, fcos, fsin, qtb, ktb, vtb);
  }
  // Flash attention
  {
    const int waves = Bc * Hc * (Sc / 16);           // 4096 waves
    attn_fwd_wmma<<<waves / AW, 32 * AW, 0, stream>>>(qtb, ktb, vtb, ctxb);
  }
  // Output GEMM: [4096,1024] x [1024,1024] + bias -> fp32 d_out
  {
    const int blocks = (M / 128) * (Dc / 64);        // 32*16 = 512
    gemm_bf16_wmma<<<blocks, 256, 0, stream>>>(ctxb, wdb, b_dense,
                                               out, nullptr, M, Dc, Dc);
  }
}
